// STFusion5_38766374813973
// MI455X (gfx1250) — matrix operand order repaired and checked by reading, not compiled, not run on hardware
//
#include <hip/hip_runtime.h>
#include <math.h>

// ---------------------------------------------------------------------------
// Problem constants (from reference): n=2, d=192, t=8, h=48, w=48
// ---------------------------------------------------------------------------
#define NN 2
#define DD 192
#define TT 8
#define HH 48
#define WW 48
#define HW (HH * WW)             // 2304
#define TOK (NN * TT * HW)       // 36864 tokens (multiple of 64)
#define UELEM ((size_t)TOK * DD) // 7,077,888 floats per d-dim tensor

typedef __attribute__((ext_vector_type(16))) __bf16 v16bf;
typedef __attribute__((ext_vector_type(8)))  float  v8f;

__device__ __forceinline__ unsigned short f32_to_bf16(float f) {
    unsigned int u = __float_as_uint(f);
    unsigned int r = u + 0x7FFFu + ((u >> 16) & 1u);   // round-to-nearest-even
    return (unsigned short)(r >> 16);
}

__device__ __forceinline__ float gelu_exact(float x) {
    return 0.5f * x * (1.0f + erff(x * 0.70710678118654752440f));
}

// ---------------------------------------------------------------------------
// WMMA GEMM:  C[M,N] = act(A[M,K] @ W + bias)
//   A:  bf16 row-major shadow (lda=K).
//   Bh: bf16 weights pre-transposed to [n][k] (ldb=K), zero-padded to
//       Npad = 64-multiple rows -> staging needs NO guards at all.
//   C (f32) and Cb (bf16 shadow) outputs optional (nullable).
//   Preconditions: M % 64 == 0, K % 32 == 0, Npad >= ceil(N,64)*64.
// Block 256 thr (8 waves), tile 64x64x32. Wave (wm 0..1, wn 0..3) owns
// rows [wm*32,+32), cols [wn*16,+16) as two stacked 16x16 accumulators.
// Staging: one async 16B global->LDS copy per thread per operand
// (global_load_async_to_lds_b128, ASYNCcnt), LDS stride 40 shorts so every
// WMMA fragment is contiguous 16B runs => ds_load_b128.
// ---------------------------------------------------------------------------
#define BM 64
#define BN 64
#define BK 32
#define LDS_STRIDE 40

template<int ACT>
__global__ __launch_bounds__(256) void wmma_gemm_k(
    const unsigned short* __restrict__ A, const unsigned short* __restrict__ Bh,
    const float* __restrict__ bias, float* __restrict__ C,
    unsigned short* __restrict__ Cb, int M, int N, int K)
{
    __shared__ unsigned short As[BM * LDS_STRIDE];
    __shared__ unsigned short Bs[BN * LDS_STRIDE];

    const int tid  = threadIdx.x;
    const int lane = tid & 31;
    const int wave = tid >> 5;
    const int wm   = wave & 1;
    const int wn   = wave >> 1;
    const int m0   = blockIdx.y * BM;
    const int n0   = blockIdx.x * BN;

    const int hf = lane >> 4;   // half-wave select
    const int ml = lane & 15;   // row (A) / col (B) within 16

    v8f acc0 = {};
    v8f acc1 = {};

    // staging: one 16B run per thread per operand (64 rows x 4 runs)
    const int sr  = tid >> 2;          // row 0..63 (A: m, B: n)
    const int srq = tid & 3;           // 16B run in row (k = 8*srq shorts)

    const unsigned short* ga = A  + (size_t)(m0 + sr) * K + 8 * srq;
    const unsigned short* gb = Bh + (size_t)(n0 + sr) * K + 8 * srq;
    const unsigned la = (unsigned)(size_t)&As[sr * LDS_STRIDE + 8 * srq];
    const unsigned lb = (unsigned)(size_t)&Bs[sr * LDS_STRIDE + 8 * srq];

    for (int k0 = 0; k0 < K; k0 += BK) {
        // ---- async stage A and B tiles (raw bf16 16B copies, ASYNCcnt) ----
        asm volatile("global_load_async_to_lds_b128 %0, %1, off"
                     :: "v"(la), "v"(ga + k0) : "memory");
        asm volatile("global_load_async_to_lds_b128 %0, %1, off"
                     :: "v"(lb), "v"(gb + k0) : "memory");
        // prefetch next K-tile (gfx1250 global_prefetch path)
        if (k0 + BK < K) {
            __builtin_prefetch(ga + k0 + BK, 0, 1);
        }
        asm volatile("s_wait_asynccnt 0" ::: "memory");
        __syncthreads();

        // ---- fragments: contiguous 16B runs -> ds_load_b128 ----
        union Frag { v16bf v; unsigned short u[16]; };
        Frag fa0, fa1, fb;
        {
            const int ra0 = (wm * 32 + ml) * LDS_STRIDE;
            const int ra1 = (wm * 32 + 16 + ml) * LDS_STRIDE;
            // A layout (ISA 7.12.2): lane holds k in [8*hf, 8*hf+8) and +16
            *(uint4*)&fa0.u[0] = *(const uint4*)&As[ra0 + 8 * hf];
            *(uint4*)&fa0.u[8] = *(const uint4*)&As[ra0 + 16 + 8 * hf];
            *(uint4*)&fa1.u[0] = *(const uint4*)&As[ra1 + 8 * hf];
            *(uint4*)&fa1.u[8] = *(const uint4*)&As[ra1 + 16 + 8 * hf];
            // B layout: lane (col=ml) holds k in [8*hf, 8*hf+8) and +16 (same as A)
            const int rb = (wn * 16 + ml) * LDS_STRIDE + 8 * hf;
            *(uint4*)&fb.u[0] = *(const uint4*)&Bs[rb];
            *(uint4*)&fb.u[8] = *(const uint4*)&Bs[rb + 16];
        }

        acc0 = __builtin_amdgcn_wmma_f32_16x16x32_bf16(
                   false, fa0.v, false, fb.v, (short)0, acc0, false, false);
        acc1 = __builtin_amdgcn_wmma_f32_16x16x32_bf16(
                   false, fa1.v, false, fb.v, (short)0, acc1, false, false);
        __syncthreads();
    }

    // ---- epilogue: C/D layout n=lane&15, M = r + 8*(lane>>4) ----
    const int gn  = n0 + wn * 16 + ml;
    const int gnc = gn < N ? gn : (N - 1);
    float bv = bias ? bias[gnc] : 0.0f;
    if (gn < N) {
        #pragma unroll
        for (int r = 0; r < 8; ++r) {
            int gm0 = m0 + wm * 32 + r + 8 * hf;   // always < M (M%64==0)
            int gm1 = gm0 + 16;
            float y0 = acc0[r] + bv;
            float y1 = acc1[r] + bv;
            if (ACT) { y0 = gelu_exact(y0); y1 = gelu_exact(y1); }
            if (C) {
                C[(size_t)gm0 * N + gn] = y0;
                C[(size_t)gm1 * N + gn] = y1;
            }
            if (Cb) {
                Cb[(size_t)gm0 * N + gn] = f32_to_bf16(y0);
                Cb[(size_t)gm1 * N + gn] = f32_to_bf16(y1);
            }
        }
    }
}

// ---------------------------------------------------------------------------
// Weight pre-conversion (once per launch; weights are < 1 MB total):
//   wcvt_id_k: W already [n][k] row-major (conv_w is (o,c)) -> bf16 copy
//   wcvt_t_k : W is (K,N) row-major -> bf16 [n][k], zero-padded to Npad rows
// ---------------------------------------------------------------------------
__global__ void wcvt_id_k(const float* __restrict__ W,
                          unsigned short* __restrict__ Wh, int total)
{
    int i = blockIdx.x * blockDim.x + threadIdx.x;
    if (i < total) Wh[i] = f32_to_bf16(W[i]);
}

__global__ void wcvt_t_k(const float* __restrict__ W,
                         unsigned short* __restrict__ Wh,
                         int K, int N, int Npad)
{
    int i = blockIdx.x * blockDim.x + threadIdx.x;
    if (i >= Npad * K) return;
    int n = i / K;
    int k = i % K;
    float v = (n < N) ? W[(size_t)k * N + n] : 0.0f;
    Wh[i] = f32_to_bf16(v);
}

// ---------------------------------------------------------------------------
// Pack conv input (bf16): A0[m, c], token m in (n,t,h,w) order, c in [0,384)
// ---------------------------------------------------------------------------
__global__ void pack_conv_a_k(const float* __restrict__ x,
                              const float* __restrict__ fuse,
                              unsigned short* __restrict__ A0)
{
    size_t idx = (size_t)blockIdx.x * blockDim.x + threadIdx.x;
    size_t total = (size_t)TOK * 384;
    if (idx >= total) return;
    int m = (int)(idx / 384);
    int c = (int)(idx % 384);
    int nn  = m / (TT * HW);
    int rem = m % (TT * HW);
    int tt  = rem / HW;
    int p   = rem % HW;
    float v;
    if (c < DD)
        v = x[(((size_t)nn * DD + c) * TT + tt) * HW + p];
    else
        v = fuse[((size_t)nn * DD + (c - DD)) * HW + p];
    A0[idx] = f32_to_bf16(v);
}

// ---------------------------------------------------------------------------
// Faithful temporal view: x_temp = xm.transpose(0,3,4,1,2).reshape(n*h*w,t,d)
// (d,t) memory reinterpreted as (t,d): channel o = tt*24 + c/8, time tq = c%8.
// Writes f32 (for LN2 residual) + bf16 shadow (GEMM A operand).
// ---------------------------------------------------------------------------
__global__ void pack_xt_k(const float* __restrict__ XM,
                          float* __restrict__ XT,
                          unsigned short* __restrict__ XTh)
{
    size_t idx = (size_t)blockIdx.x * blockDim.x + threadIdx.x;
    if (idx >= UELEM) return;
    int row = (int)(idx / DD);   // b2*8 + tt
    int c   = (int)(idx % DD);
    int b2 = row >> 3;
    int tt = row & 7;
    int nn = b2 / HW;
    int p  = b2 % HW;
    int o  = tt * (DD / TT) + (c >> 3);   // tt*24 + c/8
    int tq = c & 7;
    float v = XM[(((size_t)(nn * TT + tq) * HW) + p) * DD + o];
    XT[idx]  = v;
    XTh[idx] = f32_to_bf16(v);
}

// ---------------------------------------------------------------------------
// In-place softmax over contiguous groups of P
// ---------------------------------------------------------------------------
__global__ void softmax_groups_k(float* __restrict__ aw, int total, int P)
{
    int gid = blockIdx.x * blockDim.x + threadIdx.x;
    if (gid >= total) return;
    float* ptr = aw + (size_t)gid * P;
    float mx = ptr[0];
    for (int i = 1; i < P; ++i) mx = fmaxf(mx, ptr[i]);
    float s = 0.0f;
    for (int i = 0; i < P; ++i) { float e = __expf(ptr[i] - mx); ptr[i] = e; s += e; }
    float inv = 1.0f / s;
    for (int i = 0; i < P; ++i) ptr[i] *= inv;
}

// ---------------------------------------------------------------------------
// Spatial deformable sampling: nH=16, P=2, hd=12. bf16 output (GEMM operand).
// grid_sample semantics, zero pad; faithful ref_y<->x component swap.
// ---------------------------------------------------------------------------
__global__ void samp2d_k(const float* __restrict__ V,
                         const float* __restrict__ OFF,
                         const float* __restrict__ AW,
                         unsigned short* __restrict__ OUT)
{
    int gid = blockIdx.x * blockDim.x + threadIdx.x;
    if (gid >= TOK * 16) return;
    int m  = gid >> 4;
    int hd = gid & 15;
    int b  = m / HW;
    int l  = m % HW;
    int i  = l / WW, j = l % WW;
    float ref_y = i * (1.0f / (HH - 1));
    float ref_x = j * (1.0f / (WW - 1));

    float4 acc[3];
    #pragma unroll
    for (int u = 0; u < 3; ++u) acc[u] = make_float4(0.f, 0.f, 0.f, 0.f);

    const float* Vb = V + ((size_t)b * HW) * DD + hd * 12;

    #pragma unroll
    for (int p = 0; p < 2; ++p) {
        float o0 = OFF[(size_t)m * 64 + hd * 4 + p * 2 + 0];
        float o1 = OFF[(size_t)m * 64 + hd * 4 + p * 2 + 1];
        float a  = AW[(size_t)m * 32 + hd * 2 + p];
        float xf = (ref_y + o0 * (1.0f / WW)) * WW - 0.5f;
        float yf = (ref_x + o1 * (1.0f / HH)) * HH - 0.5f;
        float x0 = floorf(xf), y0 = floorf(yf);
        float wx = xf - x0,  wy = yf - y0;
        int x0i = (int)x0, y0i = (int)y0;
        #pragma unroll
        for (int dy = 0; dy < 2; ++dy) {
            #pragma unroll
            for (int dx = 0; dx < 2; ++dx) {
                int xi = x0i + dx, yi = y0i + dy;
                if (xi < 0 || xi >= WW || yi < 0 || yi >= HH) continue;
                float wgt = a * (dx ? wx : 1.0f - wx) * (dy ? wy : 1.0f - wy);
                const float4* vp = (const float4*)(Vb + (size_t)(yi * WW + xi) * DD);
                #pragma unroll
                for (int u = 0; u < 3; ++u) {
                    float4 vv = vp[u];
                    acc[u].x += wgt * vv.x; acc[u].y += wgt * vv.y;
                    acc[u].z += wgt * vv.z; acc[u].w += wgt * vv.w;
                }
            }
        }
    }
    ushort4* op = (ushort4*)(OUT + (size_t)m * DD + hd * 12);
    #pragma unroll
    for (int u = 0; u < 3; ++u) {
        ushort4 hv;
        hv.x = f32_to_bf16(acc[u].x); hv.y = f32_to_bf16(acc[u].y);
        hv.z = f32_to_bf16(acc[u].z); hv.w = f32_to_bf16(acc[u].w);
        op[u] = hv;
    }
}

// ---------------------------------------------------------------------------
// Temporal deformable sampling: nH=8, P=4, hd=24. bf16 output.
// ---------------------------------------------------------------------------
__global__ void samp1d_k(const float* __restrict__ V,
                         const float* __restrict__ OFF,
                         const float* __restrict__ AW,
                         unsigned short* __restrict__ OUT)
{
    int gid = blockIdx.x * blockDim.x + threadIdx.x;
    if (gid >= TOK * 8) return;
    int m2 = gid >> 3;
    int hd = gid & 7;
    int b2 = m2 >> 3;
    int l  = m2 & 7;
    float ref = l * (1.0f / (TT - 1));

    float4 acc[6];
    #pragma unroll
    for (int u = 0; u < 6; ++u) acc[u] = make_float4(0.f, 0.f, 0.f, 0.f);

    const float* Vb = V + (size_t)b2 * TT * DD + hd * 24;

    #pragma unroll
    for (int p = 0; p < 4; ++p) {
        float off = OFF[(size_t)m2 * 32 + hd * 4 + p];
        float a   = AW[(size_t)m2 * 32 + hd * 4 + p];
        float pc  = (ref + off * (1.0f / TT)) * TT - 0.5f;
        float p0  = floorf(pc);
        float wp  = pc - p0;
        int p0i = (int)p0;
        #pragma unroll
        for (int tap = 0; tap < 2; ++tap) {
            int pi = p0i + tap;
            if (pi < 0 || pi >= TT) continue;
            float wgt = a * (tap ? wp : 1.0f - wp);
            const float4* vp = (const float4*)(Vb + (size_t)pi * DD);
            #pragma unroll
            for (int u = 0; u < 6; ++u) {
                float4 vv = vp[u];
                acc[u].x += wgt * vv.x; acc[u].y += wgt * vv.y;
                acc[u].z += wgt * vv.z; acc[u].w += wgt * vv.w;
            }
        }
    }
    ushort4* op = (ushort4*)(OUT + (size_t)m2 * DD + hd * 24);
    #pragma unroll
    for (int u = 0; u < 6; ++u) {
        ushort4 hv;
        hv.x = f32_to_bf16(acc[u].x); hv.y = f32_to_bf16(acc[u].y);
        hv.z = f32_to_bf16(acc[u].z); hv.w = f32_to_bf16(acc[u].w);
        op[u] = hv;
    }
}

// ---------------------------------------------------------------------------
// LayerNorm over D=192 of (A + B), block per row (in-place safe: out may == A)
// ---------------------------------------------------------------------------
__global__ __launch_bounds__(256) void ln_residual_k(
    const float* __restrict__ A, const float* __restrict__ B,
    const float* __restrict__ g, const float* __restrict__ beta,
    float* __restrict__ out, int rows)
{
    int row = blockIdx.x;
    if (row >= rows) return;
    int c = threadIdx.x;
    __shared__ float s1[256], s2[256];
    float xv = 0.0f;
    if (c < DD) {
        xv = A[(size_t)row * DD + c];
        if (B) xv += B[(size_t)row * DD + c];
    }
    s1[c] = (c < DD) ? xv : 0.0f;
    s2[c] = (c < DD) ? xv * xv : 0.0f;
    __syncthreads();
    for (int st = 128; st > 0; st >>= 1) {
        if (c < st) { s1[c] += s1[c + st]; s2[c] += s2[c + st]; }
        __syncthreads();
    }
    float mean = s1[0] * (1.0f / DD);
    float var  = s2[0] * (1.0f / DD) - mean * mean;
    float inv  = rsqrtf(var + 1e-5f);
    if (c < DD)
        out[(size_t)row * DD + c] = (xv - mean) * inv * g[c] + beta[c];
}

// ---------------------------------------------------------------------------
// Build bf16 concat for lin1 with the faithful raw-view row permutation:
// CAT row i = (q*HW + p)*n + r  pulls buffer row j = p*t*n + q*n + r
// ---------------------------------------------------------------------------
__global__ void cat_k(const float* __restrict__ XTLN,
                      const float* __restrict__ XSLN,
                      unsigned short* __restrict__ CAT)
{
    size_t idx = (size_t)blockIdx.x * blockDim.x + threadIdx.x;
    size_t total = (size_t)TOK * 384;
    if (idx >= total) return;
    int i = (int)(idx / 384);
    int c = (int)(idx % 384);
    int r = i & 1;
    int s = i >> 1;
    int p = s % HW;
    int q = s / HW;
    int j = p * (TT * NN) + q * NN + r;   // p*16 + q*2 + r
    float v = (c < DD) ? XTLN[(size_t)j * DD + c]
                       : XSLN[(size_t)j * DD + (c - DD)];
    CAT[idx] = f32_to_bf16(v);
}

// ---------------------------------------------------------------------------
// Final: xm + x_st (gathered), LayerNorm over channels, scatter to (n,d,t,h,w)
// ---------------------------------------------------------------------------
__global__ __launch_bounds__(256) void final_ln_k(
    const float* __restrict__ XM, const float* __restrict__ H2,
    const float* __restrict__ g, const float* __restrict__ beta,
    float* __restrict__ out)
{
    int m = blockIdx.x;          // (n,t,hw) token index
    int c = threadIdx.x;
    int nn  = m / (TT * HW);
    int rem = m % (TT * HW);
    int tt  = rem / HW;
    int p   = rem % HW;
    int i   = (tt * HW + p) * NN + nn;   // x_st row

    __shared__ float s1[256], s2[256];
    float xv = 0.0f;
    if (c < DD)
        xv = XM[(size_t)m * DD + c] + H2[(size_t)i * DD + c];
    s1[c] = (c < DD) ? xv : 0.0f;
    s2[c] = (c < DD) ? xv * xv : 0.0f;
    __syncthreads();
    for (int st = 128; st > 0; st >>= 1) {
        if (c < st) { s1[c] += s1[c + st]; s2[c] += s2[c + st]; }
        __syncthreads();
    }
    float mean = s1[0] * (1.0f / DD);
    float var  = s2[0] * (1.0f / DD) - mean * mean;
    float inv  = rsqrtf(var + 1e-5f);
    if (c < DD)
        out[(((size_t)nn * DD + c) * TT + tt) * HW + p] =
            (xv - mean) * inv * g[c] + beta[c];
}

// ---------------------------------------------------------------------------
// Host launcher
// ---------------------------------------------------------------------------
extern "C" void kernel_launch(void* const* d_in, const int* in_sizes, int n_in,
                              void* d_out, int out_size, void* d_ws, size_t ws_size,
                              hipStream_t stream) {
    const float* x      = (const float*)d_in[0];
    const float* fuse   = (const float*)d_in[1];
    const float* conv_w = (const float*)d_in[2];
    const float* conv_b = (const float*)d_in[3];
    const float* s_off_w = (const float*)d_in[4];
    const float* s_off_b = (const float*)d_in[5];
    const float* s_aw_w  = (const float*)d_in[6];
    const float* s_aw_b  = (const float*)d_in[7];
    const float* s_v_w   = (const float*)d_in[8];
    const float* s_v_b   = (const float*)d_in[9];
    const float* s_o_w   = (const float*)d_in[10];
    const float* s_o_b   = (const float*)d_in[11];
    const float* t_off_w = (const float*)d_in[12];
    const float* t_off_b = (const float*)d_in[13];
    const float* t_aw_w  = (const float*)d_in[14];
    const float* t_aw_b  = (const float*)d_in[15];
    const float* t_v_w   = (const float*)d_in[16];
    const float* t_v_b   = (const float*)d_in[17];
    const float* t_o_w   = (const float*)d_in[18];
    const float* t_o_b   = (const float*)d_in[19];
    const float* ln1_g   = (const float*)d_in[20];
    const float* ln1_b   = (const float*)d_in[21];
    const float* ln2_g   = (const float*)d_in[22];
    const float* ln2_b   = (const float*)d_in[23];
    const float* ln3_g   = (const float*)d_in[24];
    const float* ln3_b   = (const float*)d_in[25];
    const float* lin1_w  = (const float*)d_in[26];
    const float* lin1_b  = (const float*)d_in[27];
    const float* lin2_w  = (const float*)d_in[28];
    const float* lin2_b  = (const float*)d_in[29];
    float* out = (float*)d_out;

    float* ws = (float*)d_ws;
    const size_t U = UELEM;
    // workspace layout (float units; peak ~6.6U ~ 178 MiB), sequential reuse:
    unsigned short* A0h = (unsigned short*)ws;              // [0,U): conv A bf16
    float* XM   = ws + U;                                   // [U,2U) persists
    unsigned short* XMh = (unsigned short*)(ws + 2 * U);    // [2U,2.5U)
    float* V    = ws + 2 * U + U / 2;                       // [2.5U,3.5U)
    float* OFF  = ws + 3 * U + U / 2;                       // [3.5U, +U/3)
    float* AW   = ws + 3 * U + U / 2 + U / 3;               // [.., +U/6) -> 4U
    unsigned short* ATTh = (unsigned short*)(ws + 4 * U);   // [4U,4.5U)
    float* SOUT = ws;                                       // reuse A0h region
    float* XSLN = ws + 4 * U + U / 2;                       // [4.5U,5.5U)
    float* XT   = ws + 5 * U + U / 2;                       // [5.5U,6.5U)
    unsigned short* XTh = (unsigned short*)(ws + 2 * U);    // reuse XMh
    float* TOUT = ws;                                       // reuse (SOUT dead)
    float* XTLN = XT;                                       // LN2 in-place
    unsigned short* CATh = (unsigned short*)ws;             // [0,U) after TOUT
    unsigned short* H1h  = (unsigned short*)(ws + 2 * U + U / 2); // reuse V
    float* H2   = ws + 3 * U;                               // [3U,4U) reuse

    // bf16 weight shadows (persist; [6.5U, 6.5U+~0.06U))
    unsigned short* wb = (unsigned short*)(ws + 6 * U + U / 2);
    size_t wo = 0;
    unsigned short* conv_wh  = wb + wo; wo += (size_t)192 * 384;
    unsigned short* s_v_wh   = wb + wo; wo += (size_t)192 * 192;
    unsigned short* s_off_wh = wb + wo; wo += (size_t)64  * 192;
    unsigned short* s_aw_wh  = wb + wo; wo += (size_t)64  * 192;
    unsigned short* s_o_wh   = wb + wo; wo += (size_t)192 * 192;
    unsigned short* t_v_wh   = wb + wo; wo += (size_t)192 * 192;
    unsigned short* t_off_wh = wb + wo; wo += (size_t)64  * 192;
    unsigned short* t_aw_wh  = wb + wo; wo += (size_t)64  * 192;
    unsigned short* t_o_wh   = wb + wo; wo += (size_t)192 * 192;
    unsigned short* lin1_wh  = wb + wo; wo += (size_t)192 * 384;
    unsigned short* lin2_wh  = wb + wo; wo += (size_t)192 * 192;

    const int T = TOK;
    const int blk = 256;
    const int gy = T / BM;            // 576 row-tiles
    float* nof = (float*)nullptr;
    unsigned short* noh = (unsigned short*)nullptr;
    #define CDIV(a) ((int)(((a) + 255) / 256))

    // 0) weight pre-conversion to bf16 [n][k] layouts (padded to 64 rows)
    wcvt_id_k<<<CDIV(192 * 384), blk, 0, stream>>>(conv_w, conv_wh, 192 * 384);
    wcvt_t_k<<<CDIV(192 * 192), blk, 0, stream>>>(s_v_w,   s_v_wh,   192, 192, 192);
    wcvt_t_k<<<CDIV(64  * 192), blk, 0, stream>>>(s_off_w, s_off_wh, 192,  64,  64);
    wcvt_t_k<<<CDIV(64  * 192), blk, 0, stream>>>(s_aw_w,  s_aw_wh,  192,  32,  64);
    wcvt_t_k<<<CDIV(192 * 192), blk, 0, stream>>>(s_o_w,   s_o_wh,   192, 192, 192);
    wcvt_t_k<<<CDIV(192 * 192), blk, 0, stream>>>(t_v_w,   t_v_wh,   192, 192, 192);
    wcvt_t_k<<<CDIV(64  * 192), blk, 0, stream>>>(t_off_w, t_off_wh, 192,  32,  64);
    wcvt_t_k<<<CDIV(64  * 192), blk, 0, stream>>>(t_aw_w,  t_aw_wh,  192,  32,  64);
    wcvt_t_k<<<CDIV(192 * 192), blk, 0, stream>>>(t_o_w,   t_o_wh,   192, 192, 192);
    wcvt_t_k<<<CDIV(192 * 384), blk, 0, stream>>>(lin1_w,  lin1_wh,  384, 192, 192);
    wcvt_t_k<<<CDIV(192 * 192), blk, 0, stream>>>(lin2_w,  lin2_wh,  192, 192, 192);

    // 1) conv 1x1 over concat(x, fuse): XM = A0 @ conv_w^T + conv_b
    pack_conv_a_k<<<CDIV((size_t)T * 384), blk, 0, stream>>>(x, fuse, A0h);
    wmma_gemm_k<0><<<dim3(3, gy), blk, 0, stream>>>(A0h, conv_wh, conv_b, XM, XMh, T, 192, 384);

    // 2) spatial branch projections
    wmma_gemm_k<0><<<dim3(3, gy), blk, 0, stream>>>(XMh, s_v_wh,   s_v_b,   V,   noh, T, 192, 192);
    wmma_gemm_k<0><<<dim3(1, gy), blk, 0, stream>>>(XMh, s_off_wh, s_off_b, OFF, noh, T,  64, 192);
    wmma_gemm_k<0><<<dim3(1, gy), blk, 0, stream>>>(XMh, s_aw_wh,  s_aw_b,  AW,  noh, T,  32, 192);
    softmax_groups_k<<<CDIV(T * 16), blk, 0, stream>>>(AW, T * 16, 2);
    samp2d_k<<<CDIV(T * 16), blk, 0, stream>>>(V, OFF, AW, ATTh);
    wmma_gemm_k<0><<<dim3(3, gy), blk, 0, stream>>>(ATTh, s_o_wh, s_o_b, SOUT, noh, T, 192, 192);
    ln_residual_k<<<T, blk, 0, stream>>>(XM, SOUT, ln1_g, ln1_b, XSLN, T);

    // 3) temporal branch (faithful raw channel/time re-view)
    pack_xt_k<<<CDIV(UELEM), blk, 0, stream>>>(XM, XT, XTh);
    wmma_gemm_k<0><<<dim3(3, gy), blk, 0, stream>>>(XTh, t_v_wh,   t_v_b,   V,   noh, T, 192, 192);
    wmma_gemm_k<0><<<dim3(1, gy), blk, 0, stream>>>(XTh, t_off_wh, t_off_b, OFF, noh, T,  32, 192);
    wmma_gemm_k<0><<<dim3(1, gy), blk, 0, stream>>>(XTh, t_aw_wh,  t_aw_b,  AW,  noh, T,  32, 192);
    softmax_groups_k<<<CDIV(T * 8), blk, 0, stream>>>(AW, T * 8, 4);
    samp1d_k<<<CDIV(T * 8), blk, 0, stream>>>(V, OFF, AW, ATTh);
    wmma_gemm_k<0><<<dim3(3, gy), blk, 0, stream>>>(ATTh, t_o_wh, t_o_b, TOUT, noh, T, 192, 192);
    ln_residual_k<<<T, blk, 0, stream>>>(XT, TOUT, ln2_g, ln2_b, XTLN, T);

    // 4) MLP over concat with raw-view row permutation
    cat_k<<<CDIV((size_t)T * 384), blk, 0, stream>>>(XTLN, XSLN, CATh);
    wmma_gemm_k<1><<<dim3(3, gy), blk, 0, stream>>>(CATh, lin1_wh, lin1_b, nof, H1h, T, 192, 384);
    wmma_gemm_k<0><<<dim3(3, gy), blk, 0, stream>>>(H1h,  lin2_wh, lin2_b, H2,  noh, T, 192, 192);

    // 5) residual + final LayerNorm, scatter to (n,d,t,h,w)
    final_ln_k<<<T, blk, 0, stream>>>(XM, H2, ln3_g, ln3_b, out);

    (void)in_sizes; (void)n_in; (void)out_size; (void)ws_size;
    #undef CDIV
}
